// GNN_90031104458804
// MI455X (gfx1250) — matrix operand order repaired and checked by reading, not compiled, not run on hardware
//
#include <hip/hip_runtime.h>
#include <stdint.h>

#define N_NODES 100000
#define N_EDGES 1600000
#define F_IN    48
#define H_DIM   64
#define NPAD    100096            // ceil(N/128)*128 : GEMM row padding

typedef __attribute__((ext_vector_type(16))) __bf16 v16bf;
typedef __attribute__((ext_vector_type(8)))  float  v8f;

union Frag { v16bf v; uint4 q[2]; uint32_t u[8]; };

// round-to-nearest-even f32 -> bf16
__device__ __forceinline__ uint16_t f2bf(float f) {
    uint32_t u = __float_as_uint(f);
    return (uint16_t)((u + 0x7FFFu + ((u >> 16) & 1u)) >> 16);
}
__device__ __forceinline__ uint32_t pack_bf16x2(float a, float b) {
    return ((uint32_t)f2bf(b) << 16) | (uint32_t)f2bf(a);
}
__device__ __forceinline__ float bf2f(uint16_t h) {
    return __uint_as_float((uint32_t)h << 16);
}

// ---------------------------------------------------------------------------
// Degree counts (shared by both SAGE layers).
// ---------------------------------------------------------------------------
__global__ void edge_count(const int* __restrict__ dst, float* __restrict__ cnt) {
    int i = blockIdx.x * blockDim.x + threadIdx.x;
    if (i < N_EDGES) atomicAdd(&cnt[dst[i]], 1.0f);
}

// ---------------------------------------------------------------------------
// Neighbor-sum scatter, f32 source (layer 1: x, 48 features).
// ---------------------------------------------------------------------------
__global__ void edge_scatter_f32(const int* __restrict__ src, const int* __restrict__ dst,
                                 const float* __restrict__ x, float* __restrict__ agg) {
    int f = threadIdx.x & 63;
    int e = blockIdx.x * 4 + (threadIdx.x >> 6);
    if (e >= N_EDGES || f >= F_IN) return;
    atomicAdd(&agg[(long)dst[e] * H_DIM + f], x[(long)src[e] * F_IN + f]);
}

// ---------------------------------------------------------------------------
// Neighbor-sum scatter, bf16 source (layer 2: h1, 64 features).
// ---------------------------------------------------------------------------
__global__ void edge_scatter_bf16(const int* __restrict__ src, const int* __restrict__ dst,
                                  const uint16_t* __restrict__ h, float* __restrict__ agg) {
    int f = threadIdx.x & 63;
    int e = blockIdx.x * 4 + (threadIdx.x >> 6);
    if (e >= N_EDGES) return;
    float v = bf2f(h[(long)src[e] * H_DIM + f]);
    atomicAdd(&agg[(long)dst[e] * H_DIM + f], v);
}

// ---------------------------------------------------------------------------
// Pack f32 rows (optionally scaled by 1/max(cnt,1) => mean) into bf16,
// zero-padded to 64 cols and NPAD rows. dst is row-major, 32 dwords/row.
// ---------------------------------------------------------------------------
__global__ void pack_rows_bf16(const float* __restrict__ srcf, int ld, int Kact,
                               const float* __restrict__ cnt, int nrows,
                               uint32_t* __restrict__ dst) {
    int idx = blockIdx.x * blockDim.x + threadIdx.x;
    if (idx >= NPAD * 32) return;
    int p   = idx & 31;
    int row = idx >> 5;
    int k   = p * 2;
    float a = 0.0f, b = 0.0f;
    if (row < nrows) {
        float s = 1.0f;
        if (cnt != nullptr) s = 1.0f / fmaxf(cnt[row], 1.0f);
        if (k     < Kact) a = srcf[(long)row * ld + k] * s;
        if (k + 1 < Kact) b = srcf[(long)row * ld + k + 1] * s;
    }
    dst[idx] = pack_bf16x2(a, b);
}

// ---------------------------------------------------------------------------
// Pre-pack weight matrix (Kact x 64 f32) into WMMA B-fragment layout
// (contiguous k, same order as A in sage_gemm, not the ISA table order; lanes 0-15 K=0..15, lanes 16-31 K=16..31,
//  VGPR v holds K = khalf*16 + 2v for column n = lane&15), zero-padded K->64.
// frag[((ntile*2 + ktile)*32 + lane)*8 + v]   (8 KB / matrix)
// ---------------------------------------------------------------------------
__global__ void prep_wfrag(const float* __restrict__ W, uint32_t* __restrict__ frag,
                           int Kact) {
    int idx = blockIdx.x * blockDim.x + threadIdx.x;
    if (idx >= 4 * 2 * 32 * 8) return;
    int v     = idx & 7;
    int lane  = (idx >> 3) & 31;
    int ktile = (idx >> 8) & 1;
    int ntile = idx >> 9;
    int n     = ntile * 16 + (lane & 15);
    int k     = ktile * 32 + (lane >> 4) * 16 + v * 2;
    float a = (k     < Kact) ? W[(long)k * H_DIM + n]       : 0.0f;
    float b = (k + 1 < Kact) ? W[(long)(k + 1) * H_DIM + n] : 0.0f;
    frag[idx] = pack_bf16x2(a, b);
}

// two contiguous b128 loads -> one 16x32 bf16 fragment
__device__ __forceinline__ v16bf load_frag(const uint32_t* __restrict__ p) {
    Frag f;
    f.q[0] = *(const uint4*)(p);
    f.q[1] = *(const uint4*)(p + 4);
    return f.v;
}

// ---------------------------------------------------------------------------
// Fused SAGE layer on pre-packed bf16 operands:
//   out = sigmoid( A1 @ W1 + bias [+ A2 @ W2] )    (A = NPAD x 64 bf16)
// 256 threads = 8 waves; wave owns 16 rows x 64 cols; A fragments hoisted
// (N-tile invariant); inner loop = B b128 loads + v_wmma_f32_16x16x32_bf16.
// Output: unguarded b16 stores into an NPAD-padded bf16 buffer.
// ---------------------------------------------------------------------------
__global__ __launch_bounds__(256)
void sage_gemm(const uint32_t* __restrict__ A1, const uint32_t* __restrict__ A2,
               const uint32_t* __restrict__ W1f, const uint32_t* __restrict__ W2f,
               const float* __restrict__ bias, uint16_t* __restrict__ outB) {
    const int lane   = threadIdx.x & 31;
    const int wave   = threadIdx.x >> 5;
    const int row0   = (blockIdx.x * 8 + wave) * 16;
    const int row    = row0 + (lane & 15);
    const int khalf4 = (lane >> 4) << 3;      // khalf * 8 dwords

    // A-fragment layout (contiguous k, matches B): lane row = lane&15,
    // pair indices ktile*16 + khalf*8 + {0..7} -> 2 x b128 each.
    const uint32_t* a1p = A1 + (long)row * 32 + khalf4;
    v16bf a1k0 = load_frag(a1p);
    v16bf a1k1 = load_frag(a1p + 16);
    v16bf a2k0 = {}, a2k1 = {};
    if (A2 != nullptr) {
        const uint32_t* a2p = A2 + (long)row * 32 + khalf4;
        a2k0 = load_frag(a2p);
        a2k1 = load_frag(a2p + 16);
    }

    #pragma unroll
    for (int ntile = 0; ntile < 4; ++ntile) {
        v8f acc = {};
        const int fb0 = (((ntile * 2 + 0) * 32) + lane) * 8;
        const int fb1 = (((ntile * 2 + 1) * 32) + lane) * 8;

        acc = __builtin_amdgcn_wmma_f32_16x16x32_bf16(
                  false, a1k0, false, load_frag(W1f + fb0), (short)0, acc, false, false);
        acc = __builtin_amdgcn_wmma_f32_16x16x32_bf16(
                  false, a1k1, false, load_frag(W1f + fb1), (short)0, acc, false, false);
        if (A2 != nullptr) {
            acc = __builtin_amdgcn_wmma_f32_16x16x32_bf16(
                      false, a2k0, false, load_frag(W2f + fb0), (short)0, acc, false, false);
            acc = __builtin_amdgcn_wmma_f32_16x16x32_bf16(
                      false, a2k1, false, load_frag(W2f + fb1), (short)0, acc, false, false);
        }

        // D layout: lanes 0-15 -> M=v, lanes 16-31 -> M=v+8, N = lane&15
        const int   n     = ntile * 16 + (lane & 15);
        const float bn    = bias[n];
        const int   mbase = (lane >> 4) * 8;
        #pragma unroll
        for (int v = 0; v < 8; ++v) {
            float val = acc[v] + bn;
            val = 1.0f / (1.0f + __expf(-val));          // sigmoid
            outB[(long)(row0 + mbase + v) * H_DIM + n] = f2bf(val);
        }
    }
}

// ---------------------------------------------------------------------------
// Head: logits = h @ Wlin2 (64x2) + b; softmax over 2 classes.
// ---------------------------------------------------------------------------
__global__ void head_softmax(const uint16_t* __restrict__ h, const float* __restrict__ W,
                             const float* __restrict__ b, float* __restrict__ out) {
    int i = blockIdx.x * blockDim.x + threadIdx.x;
    if (i >= N_NODES) return;
    float a0 = b[0], a1 = b[1];
    const uint16_t* hr = h + (long)i * H_DIM;
    #pragma unroll 8
    for (int k = 0; k < H_DIM; ++k) {
        float v = bf2f(hr[k]);
        a0 = fmaf(v, W[k * 2 + 0], a0);
        a1 = fmaf(v, W[k * 2 + 1], a1);
    }
    float mx = fmaxf(a0, a1);
    float e0 = __expf(a0 - mx);
    float e1 = __expf(a1 - mx);
    float s  = e0 + e1;
    out[(long)i * 2 + 0] = e0 / s;
    out[(long)i * 2 + 1] = e1 / s;
}

// ---------------------------------------------------------------------------
extern "C" void kernel_launch(void* const* d_in, const int* in_sizes, int n_in,
                              void* d_out, int out_size, void* d_ws, size_t ws_size,
                              hipStream_t stream) {
    const float* x     = (const float*)d_in[0];
    const int*   ei    = (const int*)d_in[1];    // edge_index (JAX default -> int32)
    const float* W1l   = (const float*)d_in[2];
    const float* b1l   = (const float*)d_in[3];
    const float* W1r   = (const float*)d_in[4];
    const float* W2l   = (const float*)d_in[5];
    const float* b2l   = (const float*)d_in[6];
    const float* W2r   = (const float*)d_in[7];
    const float* Wlin1 = (const float*)d_in[8];
    const float* blin1 = (const float*)d_in[9];
    const float* Wlin2 = (const float*)d_in[10];
    const float* blin2 = (const float*)d_in[11];
    const int* src = ei;
    const int* dst = ei + N_EDGES;

    // workspace carving (256B aligned)
    char* ws = (char*)d_ws;
    size_t off = 0;
    auto carve = [&](size_t bytes) -> char* {
        off = (off + 255) & ~(size_t)255;
        char* p = ws + off;
        off += bytes;
        return p;
    };
    float*    cnt   = (float*)carve((size_t)N_NODES * sizeof(float));
    float*    agg   = (float*)carve((size_t)N_NODES * H_DIM * sizeof(float));
    uint32_t* xbf   = (uint32_t*)carve((size_t)NPAD * 32 * sizeof(uint32_t)); // x / h2
    uint32_t* aggbf = (uint32_t*)carve((size_t)NPAD * 32 * sizeof(uint32_t)); // mean / h3
    uint32_t* h1bf  = (uint32_t*)carve((size_t)NPAD * 32 * sizeof(uint32_t));
    uint32_t* wf1l  = (uint32_t*)carve(2048 * sizeof(uint32_t));
    uint32_t* wf1r  = (uint32_t*)carve(2048 * sizeof(uint32_t));
    uint32_t* wf2l  = (uint32_t*)carve(2048 * sizeof(uint32_t));
    uint32_t* wf2r  = (uint32_t*)carve(2048 * sizeof(uint32_t));
    uint32_t* wfl1  = (uint32_t*)carve(2048 * sizeof(uint32_t));
    uint32_t* h2bf  = xbf;     // x consumed after layer-1 GEMM
    uint32_t* h3bf  = aggbf;   // mean consumed after layer-2 GEMM

    const int gemm_grid = NPAD / 128;                 // 8 waves x 16 rows / block
    const int scat_grid = (N_EDGES + 3) / 4;          // 4 edges / block
    const int pack_grid = (NPAD * 32 + 255) / 256;

    // degree counts + weight fragments + packed x (layer-independent)
    hipMemsetAsync(cnt, 0, (size_t)N_NODES * sizeof(float), stream);
    hipMemsetAsync(agg, 0, (size_t)N_NODES * H_DIM * sizeof(float), stream);
    edge_count<<<(N_EDGES + 255) / 256, 256, 0, stream>>>(dst, cnt);
    prep_wfrag<<<8, 256, 0, stream>>>(W1l,   wf1l, F_IN);
    prep_wfrag<<<8, 256, 0, stream>>>(W1r,   wf1r, F_IN);
    prep_wfrag<<<8, 256, 0, stream>>>(W2l,   wf2l, H_DIM);
    prep_wfrag<<<8, 256, 0, stream>>>(W2r,   wf2r, H_DIM);
    prep_wfrag<<<8, 256, 0, stream>>>(Wlin1, wfl1, H_DIM);
    pack_rows_bf16<<<pack_grid, 256, 0, stream>>>(x, F_IN, F_IN, nullptr, N_NODES, xbf);

    // ---- layer 1: h1 = sigmoid( mean(x) @ W1l + b1l + x @ W1r )
    edge_scatter_f32<<<scat_grid, 256, 0, stream>>>(src, dst, x, agg);
    pack_rows_bf16<<<pack_grid, 256, 0, stream>>>(agg, H_DIM, H_DIM, cnt, N_NODES, aggbf);
    sage_gemm<<<gemm_grid, 256, 0, stream>>>(aggbf, xbf, wf1l, wf1r, b1l,
                                             (uint16_t*)h1bf);

    // ---- layer 2: h2 = sigmoid( mean(h1) @ W2l + b2l + h1 @ W2r )
    hipMemsetAsync(agg, 0, (size_t)N_NODES * H_DIM * sizeof(float), stream);
    edge_scatter_bf16<<<scat_grid, 256, 0, stream>>>(src, dst, (const uint16_t*)h1bf, agg);
    pack_rows_bf16<<<pack_grid, 256, 0, stream>>>(agg, H_DIM, H_DIM, cnt, N_NODES, aggbf);
    sage_gemm<<<gemm_grid, 256, 0, stream>>>(aggbf, h1bf, wf2l, wf2r, b2l,
                                             (uint16_t*)h2bf);

    // ---- lin1: h3 = sigmoid( h2 @ Wlin1 + blin1 )
    sage_gemm<<<gemm_grid, 256, 0, stream>>>(h2bf, nullptr, wfl1, nullptr, blin1,
                                             (uint16_t*)h3bf);

    // ---- head: out = softmax( h3 @ Wlin2 + blin2 )
    head_softmax<<<(N_NODES + 255) / 256, 256, 0, stream>>>((const uint16_t*)h3bf,
                                                            Wlin2, blin2,
                                                            (float*)d_out);
}
